// OuterProductMean_67912022885144
// MI455X (gfx1250) — matrix operand order repaired and checked by reading, not compiled, not run on hardware
//
#include <hip/hip_runtime.h>

// ---- problem constants (from reference) ----
#define BDIM 2
#define SDIM 1024
#define IDIM 512
#define CDIM 32
#define EPSV 1e-5f

typedef __bf16 bf16_t;
typedef bf16_t v16bf __attribute__((ext_vector_type(16)));
typedef bf16_t v8bf  __attribute__((ext_vector_type(8)));
typedef float  v8f   __attribute__((ext_vector_type(8)));

static __device__ __forceinline__ bf16_t f2bf(float f) {
    // round-to-nearest-even fp32 -> bf16 via integer math
    unsigned u = __float_as_uint(f);
    unsigned r = (u + 0x7FFFu + ((u >> 16) & 1u)) >> 16;
    unsigned short h = (unsigned short)r;
    bf16_t o;
    __builtin_memcpy(&o, &h, 2);
    return o;
}

// -----------------------------------------------------------------------------
// Kernel 1: fused LayerNorm(channel) + dual projection (Wa, Wb), emitting bf16
// intermediates in transposed layout aT[b][c][i][s], bT[b][c][j][s] so the GEMM
// kernel gets K(=s)-contiguous operand rows. One thread per (b,s,i) row.
// -----------------------------------------------------------------------------
__global__ __launch_bounds__(256) void opm_ln_dualproj_kernel(
    const float* __restrict__ m,
    const float* __restrict__ ln_g, const float* __restrict__ ln_b,
    const float* __restrict__ Wa,   const float* __restrict__ ba,
    const float* __restrict__ Wb,   const float* __restrict__ bb,
    bf16_t* __restrict__ aT, bf16_t* __restrict__ bT)
{
    __shared__ float sWa[CDIM * CDIM];
    __shared__ float sWb[CDIM * CDIM];
    __shared__ float sg[CDIM], sbe[CDIM], sba[CDIM], sbb[CDIM];

    const int tid = threadIdx.x;
    for (int t = tid; t < CDIM * CDIM; t += 256) { sWa[t] = Wa[t]; sWb[t] = Wb[t]; }
    if (tid < CDIM) { sg[tid] = ln_g[tid]; sbe[tid] = ln_b[tid];
                      sba[tid] = ba[tid];  sbb[tid] = bb[tid]; }
    __syncthreads();

    const size_t r   = (size_t)blockIdx.x * 256 + tid;   // row over B*S*I
    const int batch  = (int)(r / ((size_t)SDIM * IDIM));
    const size_t rem = r % ((size_t)SDIM * IDIM);
    const int s = (int)(rem / IDIM);
    const int i = (int)(rem % IDIM);

    const float4* px = (const float4*)(m + r * CDIM);
    float x[CDIM];
#pragma unroll
    for (int q = 0; q < 8; ++q) {
        float4 v = px[q];
        x[4*q+0] = v.x; x[4*q+1] = v.y; x[4*q+2] = v.z; x[4*q+3] = v.w;
    }

    float mu = 0.f;
#pragma unroll
    for (int c = 0; c < CDIM; ++c) mu += x[c];
    mu *= (1.0f / CDIM);
    float var = 0.f;
#pragma unroll
    for (int c = 0; c < CDIM; ++c) { float d = x[c] - mu; var += d * d; }
    var *= (1.0f / CDIM);
    const float rs = rsqrtf(var + EPSV);

    float mh[CDIM];
#pragma unroll
    for (int c = 0; c < CDIM; ++c) mh[c] = (x[c] - mu) * rs * sg[c] + sbe[c];

    const size_t chStride = (size_t)IDIM * SDIM;
    const size_t outBase  = (size_t)batch * CDIM * chStride + (size_t)i * SDIM + s;

    for (int co = 0; co < CDIM; ++co) {
        float fa = sba[co], fb = sbb[co];
#pragma unroll
        for (int ci = 0; ci < CDIM; ++ci) {
            fa += mh[ci] * sWa[co * CDIM + ci];
            fb += mh[ci] * sWb[co * CDIM + ci];
        }
        aT[outBase + (size_t)co * chStride] = f2bf(fa);
        bT[outBase + (size_t)co * chStride] = f2bf(fb);
    }
}

// -----------------------------------------------------------------------------
// Kernel 2: per-(batch,channel) GEMM out[i,j] = sum_s aT[c][i][s]*bT[c][j][s]
// via v_wmma_f32_16x16x32_bf16, register-blocked 2x2 tiles per wave (32x32
// output region, 4 channels per wave => 128 acc VGPRs), fused /S scale and
// Wo projection epilogue through dynamic LDS.
// Block = 256 threads = 8 waves; covers one 32x32 (i,j) region, all 32 ch.
// -----------------------------------------------------------------------------
__global__ __launch_bounds__(256) void opm_wmma_gemm_kernel(
    const bf16_t* __restrict__ aT, const bf16_t* __restrict__ bT,
    const float* __restrict__ Wo,  const float* __restrict__ bo,
    float* __restrict__ out)
{
    extern __shared__ float smem[];
    float* sTile = smem;                       // [CDIM][32*32] = 32768 floats
    float* sWo   = smem + CDIM * 1024;         // 1024 floats
    float* sBo   = sWo + CDIM * CDIM;          // 32 floats

    const int tid   = threadIdx.x;
    const int lane  = tid & 31;
    const int wave  = tid >> 5;
    const int batch = blockIdx.x;
    const int i0    = blockIdx.y * 32;
    const int j0    = blockIdx.z * 32;
    const int cbase = wave * 4;

    for (int t = tid; t < CDIM * CDIM; t += 256) sWo[t] = Wo[t];
    if (tid < CDIM) sBo[tid] = bo[tid];

    const int mrow = lane & 15;   // A row / B col owned by this lane
    const int hi   = lane >> 4;   // half-wave select

    const size_t chStride = (size_t)IDIM * SDIM;

    // ISA bf16 A layout (16x32): lanes 0-15 hold K 0-7 & 16-23; lanes 16-31 hold K 8-15 & 24-31
    const bf16_t* pA = aT + ((size_t)batch * CDIM + cbase) * chStride
                          + (size_t)(i0 + mrow) * SDIM + 8 * hi;
    // ISA bf16 B layout (32x16): lanes 0-15 hold K 0-7 & 16-23; lanes 16-31 hold K 8-15 & 24-31 (col = lane&15)
    const bf16_t* pB = bT + ((size_t)batch * CDIM + cbase) * chStride
                          + (size_t)(j0 + mrow) * SDIM + 8 * hi;

    v8f acc[4][2][2];
#pragma unroll
    for (int c = 0; c < 4; ++c)
#pragma unroll
        for (int it = 0; it < 2; ++it)
#pragma unroll
            for (int jt = 0; jt < 2; ++jt)
#pragma unroll
                for (int e = 0; e < 8; ++e) acc[c][it][jt][e] = 0.f;

    for (int k0 = 0; k0 < SDIM; k0 += 32) {
#pragma unroll
        for (int c = 0; c < 4; ++c) {
            v16bf afrag[2], bfrag[2];
#pragma unroll
            for (int it = 0; it < 2; ++it) {
                const bf16_t* pa = pA + (size_t)c * chStride + (size_t)it * 16 * SDIM + k0;
                v8bf a_lo = *(const v8bf*)(pa);        // K 0-7  (+8*hi)
                v8bf a_hi = *(const v8bf*)(pa + 16);   // K 16-23 (+8*hi)
#pragma unroll
                for (int e = 0; e < 8; ++e) { afrag[it][e] = a_lo[e]; afrag[it][e + 8] = a_hi[e]; }
            }
#pragma unroll
            for (int jt = 0; jt < 2; ++jt) {
                const bf16_t* pb = pB + (size_t)c * chStride + (size_t)jt * 16 * SDIM + k0;
                v8bf b_lo = *(const v8bf*)(pb);        // K 0-7  (+8*hi)
                v8bf b_hi = *(const v8bf*)(pb + 16);   // K 16-23 (+8*hi)
#pragma unroll
                for (int e = 0; e < 8; ++e) { bfrag[jt][e] = b_lo[e]; bfrag[jt][e + 8] = b_hi[e]; }
            }
#pragma unroll
            for (int it = 0; it < 2; ++it)
#pragma unroll
                for (int jt = 0; jt < 2; ++jt)
                    acc[c][it][jt] = __builtin_amdgcn_wmma_f32_16x16x32_bf16(
                        false, afrag[it], false, bfrag[jt], (short)0, acc[c][it][jt], false, false);
        }
    }

    // stage accumulators to LDS with the 1/S mean scale
    const float scale = 1.0f / (float)SDIM;
#pragma unroll
    for (int c = 0; c < 4; ++c)
#pragma unroll
        for (int it = 0; it < 2; ++it)
#pragma unroll
            for (int jt = 0; jt < 2; ++jt)
#pragma unroll
                for (int rr = 0; rr < 8; ++rr) {
                    const int mm = 16 * it + rr + 8 * hi;      // ISA C/D layout
                    const int nn = 16 * jt + (lane & 15);
                    sTile[(cbase + c) * 1024 + mm * 32 + nn] = acc[c][it][jt][rr] * scale;
                }
    __syncthreads();

    // fused Wo projection: 4 positions per thread, C-contiguous float4 stores
    for (int rep = 0; rep < 4; ++rep) {
        const int p  = tid + rep * 256;     // 0..1023 within 32x32 region
        const int mm = p >> 5, nn = p & 31;
        float v[CDIM];
#pragma unroll
        for (int ci = 0; ci < CDIM; ++ci) v[ci] = sTile[ci * 1024 + p];

        float* op = out + ((((size_t)batch * IDIM + (i0 + mm)) * IDIM + (j0 + nn)) * CDIM);
        for (int co = 0; co < CDIM; co += 4) {
            float4 y;
            float yv[4];
#pragma unroll
            for (int q = 0; q < 4; ++q) {
                float a2 = sBo[co + q];
#pragma unroll
                for (int ci = 0; ci < CDIM; ++ci) a2 += sWo[(co + q) * CDIM + ci] * v[ci];
                yv[q] = a2;
            }
            y.x = yv[0]; y.y = yv[1]; y.z = yv[2]; y.w = yv[3];
            *(float4*)(op + co) = y;
        }
    }
}

extern "C" void kernel_launch(void* const* d_in, const int* in_sizes, int n_in,
                              void* d_out, int out_size, void* d_ws, size_t ws_size,
                              hipStream_t stream) {
    const float* m    = (const float*)d_in[0];
    const float* ln_g = (const float*)d_in[1];
    const float* ln_b = (const float*)d_in[2];
    const float* Wa   = (const float*)d_in[3];
    const float* ba   = (const float*)d_in[4];
    const float* Wb   = (const float*)d_in[5];
    const float* bb   = (const float*)d_in[6];
    const float* Wo   = (const float*)d_in[7];
    const float* bo   = (const float*)d_in[8];
    float* out = (float*)d_out;

    const size_t interElems = (size_t)BDIM * CDIM * IDIM * SDIM; // 33.5M bf16 each
    bf16_t* aT = (bf16_t*)d_ws;
    bf16_t* bT = aT + interElems;

    // Kernel 1: one thread per (b,s,i) row
    const int rows = BDIM * SDIM * IDIM;
    opm_ln_dualproj_kernel<<<rows / 256, 256, 0, stream>>>(
        m, ln_g, ln_b, Wa, ba, Wb, bb, aT, bT);

    // Kernel 2: one block per (batch, 32-wide i-tile, 32-wide j-tile)
    dim3 grid(BDIM, IDIM / 32, IDIM / 32);
    const size_t ldsBytes = (size_t)(CDIM * 1024 + CDIM * CDIM + CDIM) * sizeof(float);
    opm_wmma_gemm_kernel<<<grid, 256, ldsBytes, stream>>>(aT, bT, Wo, bo, out);
}
